// MultiHeadAttention_32238024524385
// MI455X (gfx1250) — compile-verified, matrix operand order checked by reading, not run on hardware
//
#include <hip/hip_runtime.h>
#include <hip/hip_fp16.h>
#include <stdint.h>

#define D_MODEL   1024
#define NUM_HEADS 16
#define D_K       64
#define BATCH     2
#define SEQ       2048
#define ROWS      (BATCH * SEQ)   // 4096

typedef __attribute__((ext_vector_type(16))) _Float16 v16h;
typedef __attribute__((ext_vector_type(8)))  _Float16 v8h;
typedef __attribute__((ext_vector_type(8)))  float    v8f;

// ---------------------------------------------------------------------------
// WMMA fragment loader (A operand layout; also used for B via B^T storage).
// 16-bit 16x32 fragment: lanes 0-15 hold rows 0-15, K chunks [0..7],[16..23];
// lanes 16-31 hold rows 0-15, K chunks [8..15],[24..31]. Source is K-major.
// ---------------------------------------------------------------------------
__device__ __forceinline__ v16h load_frag_kmajor(const _Float16* base, int ld, int lane) {
    const int row = lane & 15;
    const int kb  = (lane >> 4) << 3;          // 0 or 8
    const _Float16* p = base + (size_t)row * ld + kb;
    v8h lo = *reinterpret_cast<const v8h*>(p);
    v8h hi = *reinterpret_cast<const v8h*>(p + 16);
    v16h f;
#pragma unroll
    for (int i = 0; i < 8; ++i) { f[i] = lo[i]; f[i + 8] = hi[i]; }
    return f;
}

__device__ __forceinline__ v8f wmma_f16(v16h a, v16h b, v8f c) {
    return __builtin_amdgcn_wmma_f32_16x16x32_f16(false, a, false, b, (short)0, c, false, false);
}

// ---------------------------------------------------------------------------
// CDNA5 async DMA: global -> LDS, 16B per lane, tracked by ASYNCcnt.
// LDS byte address = low 32 bits of the generic pointer into __shared__.
// ---------------------------------------------------------------------------
__device__ __forceinline__ void async_copy_b128(unsigned ldsByteAddr, const _Float16* g) {
    asm volatile("global_load_async_to_lds_b128 %0, %1, off"
                 :: "v"(ldsByteAddr), "v"((unsigned long long)(uintptr_t)g)
                 : "memory");
}
__device__ __forceinline__ void wait_async0() {
    asm volatile("s_wait_asynccnt 0" ::: "memory");
}

// ---------------------------------------------------------------------------
// f32 -> f16 conversion (grid-stride)
// ---------------------------------------------------------------------------
__global__ __launch_bounds__(256) void cvt_f32_to_f16(const float* __restrict__ in,
                                                      _Float16* __restrict__ out, int n) {
    for (int i = blockIdx.x * blockDim.x + threadIdx.x; i < n; i += blockDim.x * gridDim.x)
        out[i] = (_Float16)in[i];
}

// ---------------------------------------------------------------------------
// y = A @ W^T + bias.  Block: 256 rows x 64 cols; wave: 32 rows x 64 cols.
// W k-slices (64x32, 4KB) staged cooperatively via async DMA, double-buffered.
// mode 0: f16 out [B,H,S,Dk] | mode 1: f16 out [B,H,Dk,S] | mode 2: f32 out.
// ---------------------------------------------------------------------------
__global__ __launch_bounds__(256)
void wmma_gemm_xwT(const _Float16* __restrict__ A,
                   const _Float16* __restrict__ W,
                   const float*    __restrict__ bias,
                   _Float16*       __restrict__ outH,
                   float*          __restrict__ outF,
                   int mode) {
    __shared__ __align__(16) _Float16 ldsW[2][64 * 32];   // 2 x 4KB double buffer

    const int t    = threadIdx.x;
    const int lane = t & 31;
    const int wv   = t >> 5;
    const int r0   = (blockIdx.x >> 4) * 256 + wv * 32;   // wave's first row
    const int j0   = (blockIdx.x & 15) * 64;              // block's first col

    const unsigned ldsWb[2] = { (unsigned)(uintptr_t)&ldsW[0][0],
                                (unsigned)(uintptr_t)&ldsW[1][0] };
    const int wrow = t >> 2, wch = t & 3;                 // 64 rows x 4 x 16B chunks
    const unsigned ldsOff = (unsigned)(wrow * 32 + wch * 8) * 2;
    const _Float16* wsrc = W + (size_t)(j0 + wrow) * D_MODEL + wch * 8;

    async_copy_b128(ldsWb[0] + ldsOff, wsrc);             // stage k-slice 0

    v8f acc0[4] = {}, acc1[4] = {};
    int buf = 0;
    for (int k = 0; k < D_MODEL; k += 32) {
        wait_async0();
        __syncthreads();
        if (k + 32 < D_MODEL) {
            async_copy_b128(ldsWb[buf ^ 1] + ldsOff, wsrc + (k + 32));
            __builtin_prefetch(A + (size_t)r0 * D_MODEL + k + 64, 0, 3);
        }
        v16h a0 = load_frag_kmajor(A + (size_t)r0 * D_MODEL + k, D_MODEL, lane);
        v16h a1 = load_frag_kmajor(A + (size_t)(r0 + 16) * D_MODEL + k, D_MODEL, lane);
        const _Float16* wb = &ldsW[buf][0];
#pragma unroll
        for (int n = 0; n < 4; ++n) {
            v16h b = load_frag_kmajor(wb + n * 16 * 32, 32, lane);
            acc0[n] = wmma_f16(a0, b, acc0[n]);
            acc1[n] = wmma_f16(a1, b, acc1[n]);
        }
        buf ^= 1;
    }

    // C layout: lanes 0-15 -> rows 0-7 in v[0..7]; lanes 16-31 -> rows 8-15
    const int col   = lane & 15;
    const int rbase = (lane >> 4) << 3;
#pragma unroll
    for (int half = 0; half < 2; ++half) {
        const v8f* acc = half ? acc1 : acc0;
        const int  rh  = r0 + half * 16;
#pragma unroll
        for (int n = 0; n < 4; ++n) {
            const int   j  = j0 + n * 16 + col;
            const float bj = bias[j];
#pragma unroll
            for (int r = 0; r < 8; ++r) {
                const int   row = rh + rbase + r;
                const float v   = acc[n][r] + bj;
                if (mode == 2) {
                    outF[(size_t)row * D_MODEL + j] = v;
                } else {
                    const int b_ = row / SEQ, s = row % SEQ;
                    const int h = j / D_K, dk = j % D_K;
                    if (mode == 0)
                        outH[(((size_t)b_ * NUM_HEADS + h) * SEQ + s) * D_K + dk] = (_Float16)v;
                    else
                        outH[(((size_t)b_ * NUM_HEADS + h) * D_K + dk) * SEQ + s] = (_Float16)v;
                }
            }
        }
    }
}

// ---------------------------------------------------------------------------
// Flash attention. Block = one (b,h), 8 waves x 16 queries = 128 queries.
// K block (32x64) and V^T slice (64x32) staged once per block via async DMA,
// double buffered; DMA of block i+1 overlaps compute of block i.
// ---------------------------------------------------------------------------
__global__ __launch_bounds__(256)
void wmma_flash_attn(const _Float16* __restrict__ Q,   // [B,H,S,Dk]
                     const _Float16* __restrict__ K,   // [B,H,S,Dk]
                     const _Float16* __restrict__ Vt,  // [B,H,Dk,S]
                     _Float16*       __restrict__ O) { // [B,S,D_MODEL]
    __shared__ __align__(16) _Float16 ldsK[2][32 * 64]; // 2 x 4KB
    __shared__ __align__(16) _Float16 ldsV[2][64 * 32]; // 2 x 4KB
    __shared__ __align__(16) _Float16 ldsP[8][16 * 32]; // per-wave P tile, 8KB

    const int t    = threadIdx.x;
    const int lane = t & 31;
    const int wv   = t >> 5;
    const int bh   = blockIdx.x >> 4;                   // b*NUM_HEADS + h
    const int qt   = (blockIdx.x & 15) * 8 + wv;        // wave's q-tile
    const _Float16* Qh = Q  + (size_t)bh * SEQ * D_K;
    const _Float16* Kh = K  + (size_t)bh * SEQ * D_K;
    const _Float16* Vh = Vt + (size_t)bh * D_K * SEQ;
    const int   q0    = qt * 16;
    const float scale = 0.125f;                         // 1/sqrt(64)

    const unsigned ldsKb[2] = { (unsigned)(uintptr_t)&ldsK[0][0],
                                (unsigned)(uintptr_t)&ldsK[1][0] };
    const unsigned ldsVb[2] = { (unsigned)(uintptr_t)&ldsV[0][0],
                                (unsigned)(uintptr_t)&ldsV[1][0] };
    // K block is contiguous (32*64 halfs): thread t copies 16B at t*16.
    const unsigned kOff = (unsigned)t * 16;
    // V^T slice: 64 rows of 32 halfs (64B); 4 threads per row.
    const int vrow = t >> 2, vch = t & 3;
    const unsigned vOff = (unsigned)(vrow * 32 + vch * 8) * 2;

    v16h aq[2];
#pragma unroll
    for (int c = 0; c < 2; ++c)
        aq[c] = load_frag_kmajor(Qh + (size_t)q0 * D_K + c * 32, D_K, lane);

    v8f   o[4] = {};
    float mrow[8], lrow[8];
#pragma unroll
    for (int r = 0; r < 8; ++r) { mrow[r] = -3.0e38f; lrow[r] = 0.0f; }

    _Float16* myP   = ldsP[wv];
    const int col   = lane & 15;
    const int rbase = (lane >> 4) << 3;

    // stage key-block 0
    async_copy_b128(ldsKb[0] + kOff, Kh + t * 8);
    async_copy_b128(ldsVb[0] + vOff, Vh + (size_t)vrow * SEQ + vch * 8);

    int buf = 0;
    for (int kb = 0; kb < SEQ; kb += 32) {
        wait_async0();
        __syncthreads();
        if (kb + 32 < SEQ) {
            async_copy_b128(ldsKb[buf ^ 1] + kOff,
                            Kh + (size_t)(kb + 32) * D_K + t * 8);
            async_copy_b128(ldsVb[buf ^ 1] + vOff,
                            Vh + (size_t)vrow * SEQ + (kb + 32) + vch * 8);
        }

        // ---- scores: S = Q(16x64) @ K_blk^T(64x32) from LDS ----
        const _Float16* kB = &ldsK[buf][0];
        v8f s[2] = {};
#pragma unroll
        for (int tt = 0; tt < 2; ++tt) {
#pragma unroll
            for (int c = 0; c < 2; ++c) {
                v16h bk = load_frag_kmajor(kB + (tt * 16) * D_K + c * 32, D_K, lane);
                s[tt] = wmma_f16(aq[c], bk, s[tt]);
            }
        }

        // ---- online softmax (C-layout rows align with VGPR index) ----
#pragma unroll
        for (int r = 0; r < 8; ++r) {
            float v0 = s[0][r] * scale;
            float v1 = s[1][r] * scale;
            float bmax = fmaxf(v0, v1);
#pragma unroll
            for (int off = 8; off; off >>= 1)
                bmax = fmaxf(bmax, __shfl_xor(bmax, off, 16));
            const float mnew  = fmaxf(mrow[r], bmax);
            const float alpha = __expf(mrow[r] - mnew);
            const float p0 = __expf(v0 - mnew);
            const float p1 = __expf(v1 - mnew);
            float psum = p0 + p1;
#pragma unroll
            for (int off = 8; off; off >>= 1)
                psum += __shfl_xor(psum, off, 16);
            lrow[r] = lrow[r] * alpha + psum;
            mrow[r] = mnew;
#pragma unroll
            for (int n = 0; n < 4; ++n) o[n][r] *= alpha;
            myP[(rbase + r) * 32 + col]      = (_Float16)p0;
            myP[(rbase + r) * 32 + 16 + col] = (_Float16)p1;
        }

        asm volatile("s_wait_dscnt 0" ::: "memory");

        // ---- O += P(16x32) @ V_blk(32x64) from LDS ----
        v16h ap = load_frag_kmajor(myP, 32, lane);
        const _Float16* vB = &ldsV[buf][0];
#pragma unroll
        for (int n = 0; n < 4; ++n) {
            v16h bv = load_frag_kmajor(vB + (n * 16) * 32, 32, lane);
            o[n] = wmma_f16(ap, bv, o[n]);
        }
        buf ^= 1;
    }

    // ---- finalize and store [B,S,D_MODEL] ----
    const int b_ = bh >> 4;
    const int h  = bh & 15;
#pragma unroll
    for (int n = 0; n < 4; ++n) {
#pragma unroll
        for (int r = 0; r < 8; ++r) {
            const int   srow = q0 + rbase + r;
            const float v    = o[n][r] / lrow[r];
            O[((size_t)b_ * SEQ + srow) * D_MODEL + h * D_K + n * 16 + col] = (_Float16)v;
        }
    }
}

// ---------------------------------------------------------------------------
// Host-side orchestration
// ---------------------------------------------------------------------------
extern "C" void kernel_launch(void* const* d_in, const int* in_sizes, int n_in,
                              void* d_out, int out_size, void* d_ws, size_t ws_size,
                              hipStream_t stream) {
    const float* x  = (const float*)d_in[0];
    const float* Wq = (const float*)d_in[1];
    const float* bq = (const float*)d_in[2];
    const float* Wk = (const float*)d_in[3];
    const float* bk = (const float*)d_in[4];
    const float* Wv = (const float*)d_in[5];
    const float* bv = (const float*)d_in[6];
    const float* Wo = (const float*)d_in[7];
    const float* bo = (const float*)d_in[8];
    float* out = (float*)d_out;

    char* ws = (char*)d_ws;
    const size_t MB = (size_t)1 << 20;
    _Float16* Xh  = (_Float16*)(ws + 0 * MB);   // 8 MB  [ROWS, D_MODEL]
    _Float16* Wqh = (_Float16*)(ws + 8 * MB);   // 2 MB
    _Float16* Wkh = (_Float16*)(ws + 10 * MB);  // 2 MB
    _Float16* Wvh = (_Float16*)(ws + 12 * MB);  // 2 MB
    _Float16* Woh = (_Float16*)(ws + 14 * MB);  // 2 MB
    _Float16* Qb  = (_Float16*)(ws + 16 * MB);  // 8 MB  [B,H,S,Dk]
    _Float16* Kb  = (_Float16*)(ws + 24 * MB);  // 8 MB  [B,H,S,Dk]
    _Float16* Vtb = (_Float16*)(ws + 32 * MB);  // 8 MB  [B,H,Dk,S]
    _Float16* Ob  = (_Float16*)(ws + 40 * MB);  // 8 MB  [ROWS, D_MODEL]

    const int nx = ROWS * D_MODEL;
    const int nw = D_MODEL * D_MODEL;

    cvt_f32_to_f16<<<2048, 256, 0, stream>>>(x,  Xh,  nx);
    cvt_f32_to_f16<<<1024, 256, 0, stream>>>(Wq, Wqh, nw);
    cvt_f32_to_f16<<<1024, 256, 0, stream>>>(Wk, Wkh, nw);
    cvt_f32_to_f16<<<1024, 256, 0, stream>>>(Wv, Wvh, nw);
    cvt_f32_to_f16<<<1024, 256, 0, stream>>>(Wo, Woh, nw);

    // GEMMs: 16 row-groups (256 rows) x 16 col-tiles = 256 blocks
    wmma_gemm_xwT<<<256, 256, 0, stream>>>(Xh, Wqh, bq, Qb,  nullptr, 0);
    wmma_gemm_xwT<<<256, 256, 0, stream>>>(Xh, Wkh, bk, Kb,  nullptr, 0);
    wmma_gemm_xwT<<<256, 256, 0, stream>>>(Xh, Wvh, bv, Vtb, nullptr, 1);

    // attention: 32 (b,h) x 16 q-groups = 512 blocks
    wmma_flash_attn<<<512, 256, 0, stream>>>(Qb, Kb, Vtb, Ob);

    // final projection straight into f32 d_out
    wmma_gemm_xwT<<<256, 256, 0, stream>>>(Ob, Woh, bo, nullptr, out, 2);
}
